// TreeAttention_68324339745150
// MI455X (gfx1250) — matrix operand order repaired and checked by reading, not compiled, not run on hardware
//
#include <hip/hip_runtime.h>
#include <hip/hip_bf16.h>
#include <math.h>

// ---------------------------------------------------------------------------
// Tree attention, fully reduced analytically:
//   g_s = exp(clip((x_s (Wk^T Wq) x_s^T + x_s (Wk^T rq)) * scale, -5, 7))
//   R   = sum_s g_s (per batch)
//   out_s = Gv/(256R) + Vsum*(R - g_s)/(256R) + v_s*(g_s/R - 1/256)
// Heavy work = two [65536x512]@[512x512] GEMMs via v_wmma_f32_16x16x32_f16.
// This revision: 64 rows per block (4 sub-strips), so each B fragment loaded
// from L2 feeds 4 WMMAs -> 4x less L2 traffic than the 16-row version.
// ---------------------------------------------------------------------------

typedef _Float16 v16h __attribute__((ext_vector_type(16)));
typedef _Float16 v8h  __attribute__((ext_vector_type(8)));
typedef _Float16 v4h  __attribute__((ext_vector_type(4)));
typedef float    v8f  __attribute__((ext_vector_type(8)));

#define DIM       512
#define SLEN      256
#define BATCH     256
#define NROWS     (SLEN * BATCH)          // 65536 flattened (s,b) rows
#define SCALE_F   0.044194173824159216f   // sqrt(1/512)
#define LDSTRIDE  520                     // halves per LDS row (16B-aligned, padded)
#define RPB       64                      // rows per block (4 x 16-row sub-strips)

// ------------------------- kernel 1: Mt = Wk^T @ Wq -------------------------
// Mt[j][d] = sum_e Wk[e][j] * Wq[e][d]   (f32, 512x512)
__global__ void mt_kernel(const float* __restrict__ Wq,
                          const float* __restrict__ Wk,
                          float* __restrict__ Mt) {
    __shared__ float sk[32][33];
    __shared__ float sq[32][33];
    const int j0 = blockIdx.y * 32, d0 = blockIdx.x * 32;
    const int tid = threadIdx.x;                 // 256 threads
    const int dd = tid & 31;
    const int j4 = (tid >> 5) * 4;
    float acc[4] = {0.f, 0.f, 0.f, 0.f};
    for (int e0 = 0; e0 < DIM; e0 += 32) {
        for (int i = tid; i < 1024; i += 256) {
            int ee = i >> 5, cc = i & 31;
            sk[ee][cc] = Wk[(size_t)(e0 + ee) * DIM + j0 + cc];
            sq[ee][cc] = Wq[(size_t)(e0 + ee) * DIM + d0 + cc];
        }
        __syncthreads();
        for (int ee = 0; ee < 32; ++ee) {
            float q = sq[ee][dd];
            #pragma unroll
            for (int z = 0; z < 4; ++z) acc[z] += sk[ee][j4 + z] * q;
        }
        __syncthreads();
    }
    #pragma unroll
    for (int z = 0; z < 4; ++z)
        Mt[(size_t)(j0 + j4 + z) * DIM + d0 + dd] = acc[z];
}

// ------------------------- kernel 2: u = Wk^T @ rq --------------------------
__global__ void u_kernel(const float* __restrict__ Wk,
                         const float* __restrict__ rq,
                         float* __restrict__ u) {
    int d = blockIdx.x * blockDim.x + threadIdx.x;
    if (d >= DIM) return;
    float a = 0.f;
    for (int e = 0; e < DIM; ++e) a += rq[e] * Wk[(size_t)e * DIM + d];
    u[d] = a;
}

// --------------- kernel 3: pack B operand into WMMA lane layout -------------
// Src is row-major Bc[n][k] (n = GEMM output col, k = reduction dim), 512x512.
// B as packed here (32x16 f16, wave32, contiguous k as in A): lane l -> n = l&15; halves i=0..15 -> k = (l>>4)*16 + i.
// Packed: [tile = jt*16+kt][lane][16 halves], jt in [0,32), kt in [0,16).
__global__ void pack_b_kernel(const float* __restrict__ src,
                              _Float16* __restrict__ dst) {
    int t = blockIdx.x * blockDim.x + threadIdx.x;   // 32*16*32 = 16384
    if (t >= 32 * 16 * 32) return;
    int lane = t & 31;
    int kt   = (t >> 5) & 15;
    int jt   = t >> 9;
    int n  = jt * 16 + (lane & 15);
    int k0 = kt * 32 + (lane >> 4) * 16;
    const float* s = src + (size_t)n * DIM + k0;
    _Float16* o = dst + (size_t)t * 16;              // == (tileIdx*32 + lane)*16
    #pragma unroll
    for (int i = 0; i < 16; ++i) o[i] = (_Float16)s[i];
}

// ---------------- A fragment: contiguous-k 16-bit A layout from LDS (matches B pack) ------------------
// lane<16: row=lane,   K = kt*32 + {0..15}
// lane>=16: row=lane-16, K = kt*32 + {16..31}
__device__ inline v16h load_a_frag(const _Float16* xsh, int lane, int kt) {
    int row  = lane & 15;
    int base = kt * 32 + ((lane >> 4) << 4);
    v8h lo = *(const v8h*)(xsh + row * LDSTRIDE + base);
    v8h hi = *(const v8h*)(xsh + row * LDSTRIDE + base + 8);
    v16h a;
    #pragma unroll
    for (int i = 0; i < 8; ++i) { a[i] = lo[i]; a[i + 8] = hi[i]; }
    return a;
}

// ------------------- kernel 4: fused WMMA GEMMs per 64-row strip ------------
// Computes z = x @ Mt^T (consumed on the fly -> g) and v = x @ Wv^T (stored f16).
// 4 waves; wave w owns n-tiles jt = w*8..w*8+7 for ALL four 16-row sub-strips,
// so one B fragment load feeds 4 WMMAs.
__global__ __launch_bounds__(128)
void gemm_kernel(const float* __restrict__ x,
                 const _Float16* __restrict__ bq_pack,
                 const _Float16* __restrict__ bv_pack,
                 const float* __restrict__ u,
                 _Float16* __restrict__ vout,
                 float* __restrict__ gout) {
    __shared__ _Float16 xsh[RPB * LDSTRIDE];
    __shared__ float diag_sh[RPB];

    const int tid = threadIdx.x;                 // 128 threads, 4 waves
    const int r0  = blockIdx.x * RPB;            // 64-row strip (fixed s)
    const int wave = tid >> 5, lane = tid & 31;

    // Stage x strip, f32 -> f16, into LDS (coalesced float4 loads).
    for (int i = tid; i < RPB * 128; i += 128) {
        int row = i >> 7, c4 = i & 127;
        float4 f = ((const float4*)(x + (size_t)(r0 + row) * DIM))[c4];
        v4h h; h[0] = (_Float16)f.x; h[1] = (_Float16)f.y;
               h[2] = (_Float16)f.z; h[3] = (_Float16)f.w;
        *(v4h*)(&xsh[row * LDSTRIDE + c4 * 4]) = h;
    }
    for (int i = tid; i < RPB; i += 128) diag_sh[i] = 0.f;
    __syncthreads();

    const int mbase = (lane >> 4) * 8;

    // ---- GEMM 1: z tiles, dotted with x on the fly (u folded once per n) ----
    float dacc[4][8];
    #pragma unroll
    for (int ss = 0; ss < 4; ++ss)
        #pragma unroll
        for (int g = 0; g < 8; ++g) dacc[ss][g] = 0.f;

    #pragma unroll 1
    for (int jj = 0; jj < 8; ++jj) {
        int jt = wave * 8 + jj;
        v8f c[4];
        #pragma unroll
        for (int ss = 0; ss < 4; ++ss) c[ss] = (v8f){};
        #pragma unroll
        for (int kt = 0; kt < 16; ++kt) {
            v16h b = *(const v16h*)(bq_pack + ((size_t)(jt * 16 + kt) * 32 + lane) * 16);
            #pragma unroll
            for (int ss = 0; ss < 4; ++ss) {
                v16h a = load_a_frag(xsh + ss * 16 * LDSTRIDE, lane, kt);
                c[ss] = __builtin_amdgcn_wmma_f32_16x16x32_f16(false, a, false, b,
                                                               (short)0, c[ss], false, false);
            }
        }
        int nabs = jt * 16 + (lane & 15);
        float uv = u[nabs];
        #pragma unroll
        for (int ss = 0; ss < 4; ++ss)
            #pragma unroll
            for (int g = 0; g < 8; ++g)
                dacc[ss][g] += (float)xsh[(ss * 16 + mbase + g) * LDSTRIDE + nabs]
                               * (c[ss][g] + uv);
    }
    // reduce each dacc over its 16-lane group, then merge waves via LDS atomics
    #pragma unroll
    for (int ss = 0; ss < 4; ++ss) {
        #pragma unroll
        for (int g = 0; g < 8; ++g) {
            float v = dacc[ss][g];
            v += __shfl_xor(v, 1, 32);
            v += __shfl_xor(v, 2, 32);
            v += __shfl_xor(v, 4, 32);
            v += __shfl_xor(v, 8, 32);
            if ((lane & 15) == 0) atomicAdd(&diag_sh[ss * 16 + mbase + g], v);
        }
    }

    // ---- GEMM 2: v = x @ Wv^T, stored as f16 ----
    #pragma unroll 1
    for (int jj = 0; jj < 8; ++jj) {
        int jt = wave * 8 + jj;
        v8f c[4];
        #pragma unroll
        for (int ss = 0; ss < 4; ++ss) c[ss] = (v8f){};
        #pragma unroll
        for (int kt = 0; kt < 16; ++kt) {
            v16h b = *(const v16h*)(bv_pack + ((size_t)(jt * 16 + kt) * 32 + lane) * 16);
            #pragma unroll
            for (int ss = 0; ss < 4; ++ss) {
                v16h a = load_a_frag(xsh + ss * 16 * LDSTRIDE, lane, kt);
                c[ss] = __builtin_amdgcn_wmma_f32_16x16x32_f16(false, a, false, b,
                                                               (short)0, c[ss], false, false);
            }
        }
        int n = jt * 16 + (lane & 15);
        #pragma unroll
        for (int ss = 0; ss < 4; ++ss)
            #pragma unroll
            for (int g = 0; g < 8; ++g)
                vout[(size_t)(r0 + ss * 16 + mbase + g) * DIM + n] = (_Float16)c[ss][g];
    }

    __syncthreads();
    if (tid < RPB) {
        float d = diag_sh[tid] * SCALE_F;
        d = fminf(fmaxf(d, -5.0f), 7.0f);
        gout[r0 + tid] = expf(d);
    }
}

// ------------------- kernel 5: per-batch reductions R, Vsum, Gv -------------
__global__ void reduce_kernel(const _Float16* __restrict__ vh,
                              const float* __restrict__ g,
                              float* __restrict__ Vs,
                              float* __restrict__ Gv,
                              float* __restrict__ R) {
    int b = blockIdx.x, d = threadIdx.x;         // 256 blocks x 512 threads
    float vs = 0.f, gv = 0.f, r = 0.f;
    for (int s = 0; s < SLEN; ++s) {
        float gg = g[s * BATCH + b];
        float vv = (float)vh[((size_t)(s * BATCH + b)) * DIM + d];
        vs += vv; gv += gg * vv; r += gg;
    }
    Vs[(size_t)b * DIM + d] = vs;
    Gv[(size_t)b * DIM + d] = gv;
    if (d == 0) R[b] = r;
}

// ------------------- kernel 6: elementwise combine -> out[b][s][d] ----------
__global__ void combine_kernel(const _Float16* __restrict__ vh,
                               const float* __restrict__ g,
                               const float* __restrict__ Vs,
                               const float* __restrict__ Gv,
                               const float* __restrict__ R,
                               float* __restrict__ out) {
    size_t o = (size_t)blockIdx.x * 256 + threadIdx.x;
    int d = (int)(o & 511);
    int s = (int)((o >> 9) & 255);
    int b = (int)(o >> 17);
    float gg   = g[s * BATCH + b];
    float Rb   = R[b];
    float invR = 1.0f / Rb;
    float c0   = invR * (1.0f / 256.0f);
    float vv   = (float)vh[((size_t)(s * BATCH + b)) * DIM + d];
    out[o] = Gv[(size_t)b * DIM + d] * c0
           + Vs[(size_t)b * DIM + d] * ((Rb - gg) * c0)
           + vv * (gg * invR - (1.0f / 256.0f));
}

// ---------------------------------------------------------------------------
extern "C" void kernel_launch(void* const* d_in, const int* in_sizes, int n_in,
                              void* d_out, int out_size, void* d_ws, size_t ws_size,
                              hipStream_t stream) {
    const float* x  = (const float*)d_in[0];   // [256,256,512] (s,b,d)
    const float* Wq = (const float*)d_in[1];   // [512,512]
    const float* Wk = (const float*)d_in[2];
    const float* Wv = (const float*)d_in[3];
    const float* rq = (const float*)d_in[4];   // [512]
    float* out = (float*)d_out;                // [256,256,512] (b,s,d)

    // workspace layout (all offsets 256B aligned)
    char* ws = (char*)d_ws;
    constexpr size_t OFF_MT = 0;                               // 512*512 f32 = 1 MB
    constexpr size_t OFF_U  = OFF_MT + 512 * 512 * 4;          // 512 f32
    constexpr size_t OFF_BQ = OFF_U  + 2048;                   // packed f16 512KB
    constexpr size_t OFF_BV = OFF_BQ + 512 * 512 * 2;          // packed f16 512KB
    constexpr size_t OFF_V  = OFF_BV + 512 * 512 * 2;          // 65536*512 f16 = 64 MB
    constexpr size_t OFF_G  = OFF_V  + (size_t)NROWS * DIM * 2;// 65536 f32
    constexpr size_t OFF_R  = OFF_G  + NROWS * 4;              // 256 f32
    constexpr size_t OFF_VS = OFF_R  + 1024;                   // 256*512 f32
    constexpr size_t OFF_GV = OFF_VS + 256 * 512 * 4;          // 256*512 f32

    float*    ws_Mt = (float*)(ws + OFF_MT);
    float*    ws_u  = (float*)(ws + OFF_U);
    _Float16* ws_Bq = (_Float16*)(ws + OFF_BQ);
    _Float16* ws_Bv = (_Float16*)(ws + OFF_BV);
    _Float16* ws_v  = (_Float16*)(ws + OFF_V);
    float*    ws_g  = (float*)(ws + OFF_G);
    float*    ws_R  = (float*)(ws + OFF_R);
    float*    ws_Vs = (float*)(ws + OFF_VS);
    float*    ws_Gv = (float*)(ws + OFF_GV);

    // 1) Mt = Wk^T @ Wq  (f32, tiny)
    mt_kernel<<<dim3(16, 16), 256, 0, stream>>>(Wq, Wk, ws_Mt);
    // 2) u = Wk^T @ rq
    u_kernel<<<2, 256, 0, stream>>>(Wk, rq, ws_u);
    // 3) pack B operands into WMMA lane layout (Mt rows and Wv rows are both Bc[n][k])
    pack_b_kernel<<<64, 256, 0, stream>>>(ws_Mt, ws_Bq);
    pack_b_kernel<<<64, 256, 0, stream>>>(Wv, ws_Bv);
    // 4) fused WMMA GEMMs: g (via z.x dot) and v
    gemm_kernel<<<NROWS / RPB, 128, 0, stream>>>(x, ws_Bq, ws_Bv, ws_u, ws_v, ws_g);
    // 5) per-batch reductions
    reduce_kernel<<<BATCH, DIM, 0, stream>>>(ws_v, ws_g, ws_Vs, ws_Gv, ws_R);
    // 6) combine
    combine_kernel<<<(size_t)BATCH * SLEN * DIM / 256, 256, 0, stream>>>(
        ws_v, ws_g, ws_Vs, ws_Gv, ws_R, out);
}
